// SelfLayer_7791070675480
// MI455X (gfx1250) — matrix operand order repaired and checked by reading, not compiled, not run on hardware
//
#include <hip/hip_runtime.h>
#include <cstdint>
#include <cstddef>

// ---------------- problem constants (from reference) ----------------
#define NN   32768      // nodes
#define MULC 128        // multiplicity
#define DD   1152       // 9*MUL
#define HH   384        // 3*MUL

// ---------------- WMMA types (gfx1250, wave32) ----------------
typedef __attribute__((ext_vector_type(16))) __bf16 v16bf;
typedef __attribute__((ext_vector_type(8)))  float  v8f;

union Frag { v16bf v; uint4 q[2]; };

__device__ __forceinline__ __bf16 tobf(float f) {
    uint32_t x = __builtin_bit_cast(uint32_t, f);
    uint32_t r = (x + 0x7fffu + ((x >> 16) & 1u)) >> 16;   // RNE
    uint16_t h = (uint16_t)r;
    return __builtin_bit_cast(__bf16, h);
}
__device__ __forceinline__ float frombf(__bf16 b) {
    uint16_t h = __builtin_bit_cast(uint16_t, b);
    uint32_t x = ((uint32_t)h) << 16;
    return __builtin_bit_cast(float, x);
}

// ---------------- Wigner-3j constants (real basis, derived analytically) ----
#define INV_SQRT3  0.5773502691896258f
#define INV_SQRT5  0.4472135954999579f
#define INV_SQRT6  0.4082482904638630f
#define SQRT3      1.7320508075688772f
#define C112       0.3162277660168379f   // 1/sqrt(10)
#define E112       0.1825741858350554f   // C112/sqrt(3) = 1/sqrt(30)
#define K122       0.1825741858350554f   // 1/sqrt(30)
#define A222       0.2070196678027063f   // sqrt(3/70)
#define B222       0.1195228609334394f   // 1/sqrt(70)
#define TB222      0.2390457218668787f   // 2/sqrt(70)
#define S0SC       0.5773502691896258f   // sqrt(1)/sqrt(3)
#define S1SC       0.7071067811865476f   // sqrt(3)/sqrt(6)
#define S2SC       0.9128709291752769f   // sqrt(5)/sqrt(6)
#define INV_SQRT_MUL 0.0883883476483184f // 1/sqrt(128)

// LDS row padding: +8 bf16 -> row stride 68 dwords (68 % 64 == 4 banks skew)
#define HHP  (HH + 8)     // 392
#define O3P  (MULC + 8)   // 136

// =====================================================================
// K0: pack weights into WMMA B-fragment order (bf16).
//   B layout per ISA: lane<16 -> N=lane,  elem e -> K=(e>>3)*16+(e&7)
//                     lane>=16 -> N=lane-16, elem e -> K=(e>>3)*16+8+(e&7)
// =====================================================================
struct PackArgs {
    const float* mlp[6];   // gl_w1, gl_w2, gr_w1, gr_w2, gp_w1, gp_w2 (HHxHH, used transposed)
    const float* o3[3];    // Wl, Wr, Wp  (each 3 x 128 x 128, B = W[u][v], scaled 1/sqrt(128))
};
__global__ void k_pack(PackArgs pa, __bf16* __restrict__ pk_mlp, __bf16* __restrict__ pk_o3) {
    int idx = blockIdx.x * blockDim.x + threadIdx.x;
    const int MLP_TOT = 6 * 12 * 24 * 512;   // 884736
    const int O3_TOT  = 9 * 4 * 8 * 512;     // 147456
    if (idx < MLP_TOT) {
        int mi  = idx / (12 * 24 * 512);
        int rem = idx % (12 * 24 * 512);
        int f = rem / 512, w = rem % 512;
        int lane = w / 16, e = w % 16;
        int k0 = f / 24, jt = f % 24;
        int K = k0 * 32 + (e >> 3) * 16 + (lane >> 4) * 8 + (e & 7);
        int j = jt * 16 + (lane & 15);
        pk_mlp[idx] = tobf(pa.mlp[mi][j * HH + K]);        // B[k][j] = W[j][k]
    } else if (idx < MLP_TOT + O3_TOT) {
        int id2 = idx - MLP_TOT;
        int om  = id2 / (4 * 8 * 512);
        int rem = id2 % (4 * 8 * 512);
        int s = om / 3, l = om % 3;
        int f = rem / 512, w = rem % 512;
        int lane = w / 16, e = w % 16;
        int k0 = f / 8, jt = f % 8;
        int K = k0 * 32 + (e >> 3) * 16 + (lane >> 4) * 8 + (e & 7);
        int v = jt * 16 + (lane & 15);
        pk_o3[id2] = tobf(pa.o3[s][l * MULC * MULC + K * MULC + v] * INV_SQRT_MUL);
    }
}

// =====================================================================
// K1: f0[n] = [x0, ||x1||, ||x2||]  (bf16)
// =====================================================================
__global__ void k_f0(const float* __restrict__ x, __bf16* __restrict__ f0) {
    int t = blockIdx.x * blockDim.x + threadIdx.x;   // NN*128 threads
    int n = t >> 7, u = t & 127;
    const float* row = x + (size_t)n * DD;
    float x0 = row[u];
    float s1 = 0.f;
#pragma unroll
    for (int i = 0; i < 3; ++i) { float v = row[128 + u * 3 + i]; s1 += v * v; }
    float s2 = 0.f;
#pragma unroll
    for (int i = 0; i < 5; ++i) { float v = row[512 + u * 5 + i]; s2 += v * v; }
    __bf16* fr = f0 + (size_t)n * HH;
    fr[u]        = tobf(x0);
    fr[128 + u]  = tobf(sqrtf(s1));
    fr[256 + u]  = tobf(sqrtf(s2));
}

// =====================================================================
// K2: MLP GEMM  out[n][j] = act( sum_k A[n][k] * W[j][k] + b[j] )
//   one block (8 waves) -> one 16-node tile; A staged once in LDS,
//   each wave computes 3 of the 24 16-wide column tiles. K=384 (12 WMMAs).
//   mode 1: SiLU + store bf16 ; mode 0: no act + store f32
// =====================================================================
__global__ void __launch_bounds__(256)
k_mlp_gemm(const __bf16* __restrict__ A, const __bf16* __restrict__ pkB,
           const float* __restrict__ bias, void* __restrict__ out, int mode) {
    __shared__ __bf16 sA[16 * HHP];                  // 16 rows, padded
    int nt = blockIdx.x;                             // 0..2047
    // cooperative stage: 16*384 bf16 = 768 x 16B chunks (rows are contiguous)
    const uint4* Aq = reinterpret_cast<const uint4*>(A + (size_t)nt * 16 * HH);
#pragma unroll
    for (int c = threadIdx.x; c < 768; c += 256) {
        int row = c / 48, cc = c % 48;
        reinterpret_cast<uint4*>(sA + row * HHP)[cc] = Aq[c];
    }
    __syncthreads();

    int wave = threadIdx.x >> 5;
    int lane = threadIdx.x & 31;
    int khalf = lane >> 4;
    int mrow  = lane & 15;
    const __bf16* arow = sA + mrow * HHP;

    for (int rep = 0; rep < 3; ++rep) {
        int jt = wave + rep * 8;                     // 0..23
        v8f acc = {0.f, 0.f, 0.f, 0.f, 0.f, 0.f, 0.f, 0.f};
#pragma unroll 4
        for (int k0 = 0; k0 < 12; ++k0) {
            Frag a, b;
            const __bf16* ap = arow + k0 * 32 + khalf * 8;          // ds_load_b128 x2
            a.q[0] = *reinterpret_cast<const uint4*>(ap);
            a.q[1] = *reinterpret_cast<const uint4*>(ap + 16);
            const __bf16* bp = pkB + ((size_t)(k0 * 24 + jt) * 32 + lane) * 16;
            b.q[0] = *reinterpret_cast<const uint4*>(bp);
            b.q[1] = *reinterpret_cast<const uint4*>(bp + 8);
            if (k0 < 11) __builtin_prefetch(bp + 24 * 32 * 16, 0, 1);  // next k0 B frag
            acc = __builtin_amdgcn_wmma_f32_16x16x32_bf16(false, a.v, false, b.v,
                                                          (short)0, acc, false, false);
        }
        int col = jt * 16 + (lane & 15);
        float bj = bias[col];
#pragma unroll
        for (int r = 0; r < 8; ++r) {
            float v = acc[r] + bj;
            int row = nt * 16 + khalf * 8 + r;       // C/D layout: lanes>=16 hold M=8..15
            if (mode) {
                v = v / (1.f + __expf(-v));          // SiLU
                ((__bf16*)out)[(size_t)row * HH + col] = tobf(v);
            } else {
                ((float*)out)[(size_t)row * HH + col] = v;
            }
        }
    }
}

// =====================================================================
// K3: gate original x into component-major bf16 A matrix (9N x 128).
//   plane p: 0 -> (l=0); 1..3 -> (l=1,i); 4..8 -> (l=2,i)
//   l=0 output is the gate itself (reference: g0 replaces x0).
// =====================================================================
__global__ void k_gate_x(const float* __restrict__ x, const float* __restrict__ g,
                         __bf16* __restrict__ dst) {
    int t = blockIdx.x * blockDim.x + threadIdx.x;   // 9*NN*128 threads
    int p = t >> 22;                                 // / (NN*128)
    int rem = t & ((NN * 128) - 1);
    int n = rem >> 7, u = rem & 127;
    int l, i, ol, d;
    if (p == 0)      { l = 0; i = 0;     ol = 0;   d = 1; }
    else if (p < 4)  { l = 1; i = p - 1; ol = 128; d = 3; }
    else             { l = 2; i = p - 4; ol = 512; d = 5; }
    float gv = g[(size_t)n * HH + l * 128 + u];
    float val = (l == 0) ? gv : x[(size_t)n * DD + ol + u * d + i] * gv;
    dst[t] = tobf(val);
}

// same, but source is bf16 component-major tensor-product output
__global__ void k_gate_tp(const __bf16* __restrict__ tp, const float* __restrict__ g,
                          __bf16* __restrict__ dst) {
    int t = blockIdx.x * blockDim.x + threadIdx.x;
    int p = t >> 22;
    int rem = t & ((NN * 128) - 1);
    int n = rem >> 7, u = rem & 127;
    int l = (p == 0) ? 0 : (p < 4 ? 1 : 2);
    float gv = g[(size_t)n * HH + l * 128 + u];
    float val = (l == 0) ? gv : frombf(tp[t]) * gv;
    dst[t] = tobf(val);
}

// =====================================================================
// K4: o3-linear GEMM over the 9N x 128 component-major A, K=128.
//   one block (8 waves) -> one 16-row tile; A staged in LDS, wave w ->
//   column tile w. Scale 1/sqrt(128) folded into packed weights; bias l=0.
//   outMid != 0  -> store bf16 component-major (xl / xr)
//   outMid == 0  -> final: scatter into d_out layout + old_fii
// =====================================================================
__global__ void __launch_bounds__(256)
k_o3_gemm(const __bf16* __restrict__ A, const __bf16* __restrict__ pkSet,
          const float* __restrict__ bias0,
          __bf16* __restrict__ outMid,
          float* __restrict__ outFinal, const float* __restrict__ oldf) {
    __shared__ __bf16 sA[16 * O3P];                  // 16 rows, padded
    int mt = blockIdx.x;                             // 0..18431
    int r0 = mt * 16;
    // cooperative stage: 16*128 bf16 = 256 x 16B chunks, one per thread
    {
        int c = threadIdx.x;
        int row = c >> 4, cc = c & 15;
        reinterpret_cast<uint4*>(sA + row * O3P)[cc] =
            reinterpret_cast<const uint4*>(A + (size_t)r0 * 128)[c];
    }
    __syncthreads();

    int wave = threadIdx.x >> 5;                     // = column tile jt (0..7)
    int lane = threadIdx.x & 31;
    int jt = wave;
    int l = (r0 < NN) ? 0 : ((r0 < 4 * NN) ? 1 : 2);
    const __bf16* pkB = pkSet + l * (4 * 8 * 512);
    int khalf = lane >> 4;
    const __bf16* arow = sA + (lane & 15) * O3P;
    v8f acc = {0.f, 0.f, 0.f, 0.f, 0.f, 0.f, 0.f, 0.f};
#pragma unroll
    for (int k0 = 0; k0 < 4; ++k0) {
        Frag a, b;
        const __bf16* ap = arow + k0 * 32 + khalf * 8;
        a.q[0] = *reinterpret_cast<const uint4*>(ap);
        a.q[1] = *reinterpret_cast<const uint4*>(ap + 16);
        const __bf16* bp = pkB + ((size_t)(k0 * 8 + jt) * 32 + lane) * 16;
        b.q[0] = *reinterpret_cast<const uint4*>(bp);
        b.q[1] = *reinterpret_cast<const uint4*>(bp + 8);
        acc = __builtin_amdgcn_wmma_f32_16x16x32_bf16(false, a.v, false, b.v,
                                                      (short)0, acc, false, false);
    }
    int col = jt * 16 + (lane & 15);
    float badd = (bias0 != nullptr && l == 0) ? bias0[col] : 0.f;
    if (outMid) {
#pragma unroll
        for (int r = 0; r < 8; ++r) {
            int rowm = r0 + khalf * 8 + r;
            outMid[(size_t)rowm * 128 + col] = tobf(acc[r] + badd);
        }
    } else {
        int p = r0 >> 15;              // plane = r0 / NN  (NN = 1<<15)
        int li, ol, d;
        if (p == 0)      { ol = 0;   d = 1; li = 0; }
        else if (p < 4)  { ol = 128; d = 3; li = p - 1; }
        else             { ol = 512; d = 5; li = p - 4; }
#pragma unroll
        for (int r = 0; r < 8; ++r) {
            int rowm = r0 + khalf * 8 + r;
            int n = rowm & (NN - 1);
            size_t idx = (size_t)n * DD + ol + col * d + li;
            outFinal[idx] = acc[r] + badd + oldf[idx];
        }
    }
}

// =====================================================================
// K5: Wigner-3j tensor product (hard-coded sparse tables) + second f0.
// =====================================================================
__device__ __forceinline__ void t1_contract(const float* v, const float* q, float wq, float* o1) {
    // 1 (x) 2 -> 1 via symmetric (1,1,2) tensor
    o1[0] += wq * (C112 * (v[2] * q[0] + v[1] * q[1] - v[0] * q[4]) - E112 * v[0] * q[2]);
    o1[1] += wq * (C112 * (v[0] * q[1] + v[2] * q[3]) + 2.f * E112 * v[1] * q[2]);
    o1[2] += wq * (C112 * (v[0] * q[0] + v[1] * q[3] + v[2] * q[4]) - E112 * v[2] * q[2]);
}
__device__ __forceinline__ void p122_contract(const float* v, const float* q, float wk, float* o2) {
    // 1 (x) 2 -> 2 (generator/antisymmetric coupling), wk includes K122
    o2[0] += wk * (-v[0] * q[1] - 2.f * v[1] * q[4] + v[2] * q[3]);
    o2[1] += wk * ( v[0] * q[0] - v[1] * q[3] + SQRT3 * v[2] * q[2] + v[2] * q[4]);
    o2[2] += wk * ( SQRT3 * (v[0] * q[3] - v[2] * q[1]));
    o2[3] += wk * (-SQRT3 * v[0] * q[2] + v[0] * q[4] + v[1] * q[1] - v[2] * q[0]);
    o2[4] += wk * (-v[0] * q[3] + 2.f * v[1] * q[0] - v[2] * q[1]);
}

__global__ void k_tp(const __bf16* __restrict__ xl, const __bf16* __restrict__ xr,
                     const float* __restrict__ tpw,
                     __bf16* __restrict__ tpout, __bf16* __restrict__ f0p) {
    int t = blockIdx.x * blockDim.x + threadIdx.x;   // NN*128 threads
    int n = t >> 7, u = t & 127;
    float L[9], R[9];
#pragma unroll
    for (int p = 0; p < 9; ++p) {
        size_t o = (size_t)p * ((size_t)NN * 128) + ((size_t)n << 7) + u;
        L[p] = frombf(xl[o]);
        R[p] = frombf(xr[o]);
    }
    float w[15];
#pragma unroll
    for (int j = 0; j < 15; ++j) w[j] = tpw[j * 128 + u];

    const float* L1 = L + 1; const float* L2 = L + 4;
    const float* R1 = R + 1; const float* R2 = R + 4;
    float o0 = 0.f, o1[3] = {0.f, 0.f, 0.f}, o2[5] = {0.f, 0.f, 0.f, 0.f, 0.f};

    // (0,0,0)
    o0 += w[0] * L[0] * R[0];
    // (0,1,1), (1,0,1)
#pragma unroll
    for (int j = 0; j < 3; ++j) {
        o1[j] += w[1] * INV_SQRT3 * L[0] * R1[j];
        o1[j] += w[3] * INV_SQRT3 * L1[j] * R[0];
    }
    // (0,2,2), (2,0,2)
#pragma unroll
    for (int j = 0; j < 5; ++j) {
        o2[j] += w[2] * INV_SQRT5 * L[0] * R2[j];
        o2[j] += w[9] * INV_SQRT5 * L2[j] * R[0];
    }
    // (1,1,0)
    o0 += w[4] * INV_SQRT3 * (L1[0] * R1[0] + L1[1] * R1[1] + L1[2] * R1[2]);
    // (1,1,1): cross product / sqrt(6)
    o1[0] += w[5] * INV_SQRT6 * (L1[1] * R1[2] - L1[2] * R1[1]);
    o1[1] += w[5] * INV_SQRT6 * (L1[2] * R1[0] - L1[0] * R1[2]);
    o1[2] += w[5] * INV_SQRT6 * (L1[0] * R1[1] - L1[1] * R1[0]);
    // (1,1,2): symmetric traceless coupling
    {
        float cw = w[6];
        o2[0] += cw * C112 * (L1[2] * R1[0] + L1[0] * R1[2]);
        o2[1] += cw * C112 * (L1[0] * R1[1] + L1[1] * R1[0]);
        o2[2] += cw * E112 * (2.f * L1[1] * R1[1] - L1[0] * R1[0] - L1[2] * R1[2]);
        o2[3] += cw * C112 * (L1[1] * R1[2] + L1[2] * R1[1]);
        o2[4] += cw * C112 * (L1[2] * R1[2] - L1[0] * R1[0]);
    }
    // (1,2,1) and (2,1,1)
    t1_contract(L1, R2, w[7],  o1);
    t1_contract(R1, L2, w[10], o1);
    // (1,2,2) and (2,1,2)
    p122_contract(L1, R2, w[8]  * K122, o2);
    p122_contract(R1, L2, w[11] * K122, o2);
    // (2,2,0)
    o0 += w[12] * INV_SQRT5 * (L2[0] * R2[0] + L2[1] * R2[1] + L2[2] * R2[2] +
                               L2[3] * R2[3] + L2[4] * R2[4]);
    // (2,2,1)
    {
        float wk = w[13] * K122;
        const float* A = L2; const float* B = R2;
        o1[0] += wk * (A[0] * B[1] - A[1] * B[0] - SQRT3 * A[2] * B[3] + SQRT3 * A[3] * B[2]
                       - A[3] * B[4] + A[4] * B[3]);
        o1[1] += wk * (2.f * A[0] * B[4] - 2.f * A[4] * B[0] + A[1] * B[3] - A[3] * B[1]);
        o1[2] += wk * (-A[0] * B[3] + A[3] * B[0] - SQRT3 * A[1] * B[2] + SQRT3 * A[2] * B[1]
                       - A[1] * B[4] + A[4] * B[1]);
    }
    // (2,2,2)
    {
        float wk = w[14];
        const float* A = L2; const float* B = R2;
        o2[0] += wk * (-A222 * (A[1] * B[3] + A[3] * B[1]) + TB222 * (A[0] * B[2] + A[2] * B[0]));
        o2[1] += wk * (-A222 * (A[0] * B[3] + A[3] * B[0]) + A222 * (A[1] * B[4] + A[4] * B[1])
                       - B222 * (A[1] * B[2] + A[2] * B[1]));
        o2[2] += wk * ( TB222 * (A[0] * B[0] + A[4] * B[4]) - B222 * (A[1] * B[1] + A[3] * B[3])
                       - TB222 * A[2] * B[2]);
        o2[3] += wk * (-A222 * (A[0] * B[1] + A[1] * B[0]) - A222 * (A[3] * B[4] + A[4] * B[3])
                       - B222 * (A[3] * B[2] + A[2] * B[3]));
        o2[4] += wk * ( A222 * A[1] * B[1] - A222 * A[3] * B[3] + TB222 * (A[4] * B[2] + A[2] * B[4]));
    }

    // per-l normalization sqrt(2l+1)/sqrt(n_paths)
    o0 *= S0SC;
#pragma unroll
    for (int j = 0; j < 3; ++j) o1[j] *= S1SC;
#pragma unroll
    for (int j = 0; j < 5; ++j) o2[j] *= S2SC;

    // store tp (component-major) and second normgate input f0p
    size_t base = ((size_t)n << 7) + u;
    const size_t PL = (size_t)NN * 128;
    tpout[base] = tobf(o0);
#pragma unroll
    for (int j = 0; j < 3; ++j) tpout[(size_t)(1 + j) * PL + base] = tobf(o1[j]);
#pragma unroll
    for (int j = 0; j < 5; ++j) tpout[(size_t)(4 + j) * PL + base] = tobf(o2[j]);

    float s1 = o1[0] * o1[0] + o1[1] * o1[1] + o1[2] * o1[2];
    float s2 = o2[0] * o2[0] + o2[1] * o2[1] + o2[2] * o2[2] + o2[3] * o2[3] + o2[4] * o2[4];
    __bf16* fr = f0p + (size_t)n * HH;
    fr[u]       = tobf(o0);
    fr[128 + u] = tobf(sqrtf(s1));
    fr[256 + u] = tobf(sqrtf(s2));
}

// =====================================================================
// launcher
// =====================================================================
extern "C" void kernel_launch(void* const* d_in, const int* in_sizes, int n_in,
                              void* d_out, int out_size, void* d_ws, size_t ws_size,
                              hipStream_t stream) {
    (void)in_sizes; (void)n_in; (void)out_size; (void)ws_size;
    const float* x     = (const float*)d_in[0];
    const float* oldf  = (const float*)d_in[1];
    const float* gl_w1 = (const float*)d_in[2];
    const float* gl_b1 = (const float*)d_in[3];
    const float* gl_w2 = (const float*)d_in[4];
    const float* gl_b2 = (const float*)d_in[5];
    const float* gr_w1 = (const float*)d_in[6];
    const float* gr_b1 = (const float*)d_in[7];
    const float* gr_w2 = (const float*)d_in[8];
    const float* gr_b2 = (const float*)d_in[9];
    const float* gp_w1 = (const float*)d_in[10];
    const float* gp_b1 = (const float*)d_in[11];
    const float* gp_w2 = (const float*)d_in[12];
    const float* gp_b2 = (const float*)d_in[13];
    const float* Wl    = (const float*)d_in[14];
    const float* bl    = (const float*)d_in[15];
    const float* Wr    = (const float*)d_in[16];
    const float* br    = (const float*)d_in[17];
    const float* Wp    = (const float*)d_in[18];
    const float* tpw   = (const float*)d_in[19];
    float* out = (float*)d_out;

    // workspace carve-out (~380 MB), 256B aligned
    char* ws = (char*)d_ws;
    size_t off = 0;
    auto take = [&](size_t bytes) -> char* {
        char* p = ws + off;
        off = (off + bytes + 255) & ~(size_t)255;
        return p;
    };
    __bf16* pk_mlp = (__bf16*)take((size_t)884736 * 2);
    __bf16* pk_o3  = (__bf16*)take((size_t)147456 * 2);
    __bf16* f0     = (__bf16*)take((size_t)NN * HH * 2);       // f0, later f0'
    __bf16* hbuf   = (__bf16*)take((size_t)NN * HH * 2);       // MLP hidden
    float*  gA     = (float*) take((size_t)NN * HH * 4);       // gL, later gP
    float*  gB     = (float*) take((size_t)NN * HH * 4);       // gR
    __bf16* xg     = (__bf16*)take((size_t)9 * NN * 128 * 2);  // gated A, later tp
    __bf16* xlb    = (__bf16*)take((size_t)9 * NN * 128 * 2);  // xl, later gated tp
    __bf16* xrb    = (__bf16*)take((size_t)9 * NN * 128 * 2);  // xr

    PackArgs pa;
    pa.mlp[0] = gl_w1; pa.mlp[1] = gl_w2; pa.mlp[2] = gr_w1;
    pa.mlp[3] = gr_w2; pa.mlp[4] = gp_w1; pa.mlp[5] = gp_w2;
    pa.o3[0] = Wl; pa.o3[1] = Wr; pa.o3[2] = Wp;

    const size_t MS = 147456;     // bf16 elems per packed MLP matrix
    const int MLP_BLK  = 2048;    // one block per 16-node tile (8 waves x 3 jt)
    const int O3_BLK   = 18432;   // one block per 16-row tile (8 waves = 8 jt)
    const int EW_BLK   = 147456;  // 9N*128 / 256
    const int NU_BLK   = 16384;   // N*128 / 256

    k_pack<<<4032, 256, 0, stream>>>(pa, pk_mlp, pk_o3);
    k_f0<<<NU_BLK, 256, 0, stream>>>(x, f0);

    // gl / gr gate MLPs
    k_mlp_gemm<<<MLP_BLK, 256, 0, stream>>>(f0,   pk_mlp + 0 * MS, gl_b1, hbuf, 1);
    k_mlp_gemm<<<MLP_BLK, 256, 0, stream>>>(hbuf, pk_mlp + 1 * MS, gl_b2, gA,   0);
    k_mlp_gemm<<<MLP_BLK, 256, 0, stream>>>(f0,   pk_mlp + 2 * MS, gr_b1, hbuf, 1);
    k_mlp_gemm<<<MLP_BLK, 256, 0, stream>>>(hbuf, pk_mlp + 3 * MS, gr_b2, gB,   0);

    // gate + o3-linear (Wl, Wr)
    k_gate_x<<<EW_BLK, 256, 0, stream>>>(x, gA, xg);
    k_o3_gemm<<<O3_BLK, 256, 0, stream>>>(xg, pk_o3 + 0 * 3 * 16384, bl, xlb, nullptr, nullptr);
    k_gate_x<<<EW_BLK, 256, 0, stream>>>(x, gB, xg);
    k_o3_gemm<<<O3_BLK, 256, 0, stream>>>(xg, pk_o3 + 1 * 3 * 16384, br, xrb, nullptr, nullptr);

    // tensor product -> tp (into xg), f0' (into f0)
    k_tp<<<NU_BLK, 256, 0, stream>>>(xlb, xrb, tpw, xg, f0);

    // gp gate MLP
    k_mlp_gemm<<<MLP_BLK, 256, 0, stream>>>(f0,   pk_mlp + 4 * MS, gp_b1, hbuf, 1);
    k_mlp_gemm<<<MLP_BLK, 256, 0, stream>>>(hbuf, pk_mlp + 5 * MS, gp_b2, gA,   0);

    // gate tp, final o3-linear (Wp) fused with merge + old_fii add
    k_gate_tp<<<EW_BLK, 256, 0, stream>>>(xg, gA, xlb);
    k_o3_gemm<<<O3_BLK, 256, 0, stream>>>(xlb, pk_o3 + 2 * 3 * 16384, nullptr, nullptr, out, oldf);
}
